// MambaBlock_79714593014155
// MI455X (gfx1250) — matrix operand order repaired and checked by reading, not compiled, not run on hardware
//
#include <hip/hip_runtime.h>
#include <stdint.h>

typedef __bf16 bf16_t;
typedef __attribute__((ext_vector_type(16))) __bf16 v16bf;
typedef __attribute__((ext_vector_type(8)))  __bf16 v8bf;
typedef __attribute__((ext_vector_type(8)))  float   v8f;

#define D_MODEL 1024
#define D_INNER 2048
#define D_STATE 16
#define D_CONVK 4
#define DT_RANK 64
#define BATCH   4
#define SEQLEN  2048
#define NTOK    (BATCH * SEQLEN)         // 8192 tokens
#define PROJ_W  (DT_RANK + 2 * D_STATE)  // 96

// ---------- fp32 -> bf16 (round to nearest even) ----------
__device__ __forceinline__ bf16_t f2bf(float f) {
    union { float f; uint32_t u; } c; c.f = f;
    uint32_t r = c.u + 0x7FFFu + ((c.u >> 16) & 1u);
    uint16_t h = (uint16_t)(r >> 16);
    return __builtin_bit_cast(bf16_t, h);
}

__device__ __forceinline__ float siluf(float v) {
    return v / (1.0f + __expf(-v));
}

// ---------- CDNA5 async copy: one 16B chunk global -> LDS (ASYNCcnt) ----------
__device__ __forceinline__ void async_cp16(uint32_t lds_addr, const char* gaddr) {
    asm volatile("global_load_async_to_lds_b128 %0, %1, off"
                 :: "v"(lds_addr), "v"(gaddr)
                 : "memory");
}

// ---------- weight conversion ----------
__global__ void k_cvt_bf16(const float* __restrict__ src, bf16_t* __restrict__ dst, int n) {
    int i = blockIdx.x * blockDim.x + threadIdx.x;
    if (i < n) dst[i] = f2bf(src[i]);
}

// ---------- LayerNorm per token, emit bf16 ----------
__global__ __launch_bounds__(256)
void k_layernorm(const float* __restrict__ x, const float* __restrict__ g,
                 const float* __restrict__ b, bf16_t* __restrict__ h) {
    const int tok = blockIdx.x;
    const float* xr = x + (size_t)tok * D_MODEL;
    float v0[4];
    float s = 0.f, s2 = 0.f;
#pragma unroll
    for (int j = 0; j < 4; ++j) {
        float t = xr[threadIdx.x + j * 256];
        v0[j] = t; s += t; s2 += t * t;
    }
    __shared__ float rs[256], rq[256];
    rs[threadIdx.x] = s; rq[threadIdx.x] = s2;
    __syncthreads();
    for (int st = 128; st > 0; st >>= 1) {
        if (threadIdx.x < st) {
            rs[threadIdx.x] += rs[threadIdx.x + st];
            rq[threadIdx.x] += rq[threadIdx.x + st];
        }
        __syncthreads();
    }
    float mu  = rs[0] * (1.0f / D_MODEL);
    float var = rq[0] * (1.0f / D_MODEL) - mu * mu;
    float inv = rsqrtf(var + 1e-5f);
    bf16_t* hr = h + (size_t)tok * D_MODEL;
#pragma unroll
    for (int j = 0; j < 4; ++j) {
        int i = threadIdx.x + j * 256;
        hr[i] = f2bf((v0[j] - mu) * inv * g[i] + b[i]);
    }
}

// ---------- bf16 WMMA GEMM: C[M,N] = A[M,K] @ B[N,K]^T (+ residual) ----------
// Block: 256 threads = 8 waves; block tile 64(M) x 128(N); wave tile 16 x 64.
// K staged 64-deep per LDS buffer via GLOBAL_LOAD_ASYNC_TO_LDS_B128, double
// buffered (ASYNCcnt software pipeline). Row slabs padded 128B -> 144B so the
// 16-lane ds_load_b128 fragment reads are bank-conflict free.
// Fragment layouts per CDNA5 ISA 7.12.2 (wave32):
//   A 16x32 bf16: lane half g holds K = [kb+8g .. +7] and [kb+16+8g .. +7]
//   B 32x16 bf16: lane half g (column = lane&15) holds K = [kb+8g .. +7] and [kb+16+8g .. +7]
//   C/D f32 16x16: VGPR r, lanes0-15 -> M=r, N=lane; lanes16-31 -> M=8+r, N=lane-16
#define KDEPTH       64                      // K elements per LDS buffer
#define SLAB_BYTES   (KDEPTH * 2)            // 128B of K per row
#define ROW_STRIDE   (SLAB_BYTES + 16)       // 144B padded
#define A_BUF_BYTES  (64  * ROW_STRIDE)      // 9216
#define B_BUF_BYTES  (128 * ROW_STRIDE)      // 18432
#define BUF_BYTES    (A_BUF_BYTES + B_BUF_BYTES)   // 27648 (x2 = 54KB LDS)

__global__ __launch_bounds__(256)
void k_gemm_bf16(const bf16_t* __restrict__ A, const bf16_t* __restrict__ Bm,
                 const float* __restrict__ Res, float* __restrict__ C,
                 int M, int N, int K) {
    __shared__ alignas(16) unsigned char smem[2 * BUF_BYTES];
    const int tid  = threadIdx.x;
    const int lane = tid & 31;
    const int wave = tid >> 5;
    const int wm = wave & 3;                 // 4 waves along M
    const int wn = wave >> 2;                // 2 wave groups along N
    const int mBlock = blockIdx.y * 64;
    const int nBlock = blockIdx.x * 128;
    const int g   = lane >> 4;               // half-wave group
    const int r16 = lane & 15;

    // --- staging assignment: 8 x 16B chunks per 128B row slab ---
    // A tile: 64 rows * 8 chunks = 512 -> 2 per thread
    // B tile: 128 rows * 8 chunks = 1024 -> 4 per thread
    const uint32_t ldsBase = (uint32_t)(uintptr_t)&smem[0];
    const char* gsrc[6];
    uint32_t    ldst[6];
#pragma unroll
    for (int q = 0; q < 2; ++q) {
        int idx = tid + q * 256;
        int row = idx >> 3, ch = (idx & 7) * 16;
        gsrc[q] = (const char*)A + ((size_t)(mBlock + row) * K) * 2 + ch;
        ldst[q] = ldsBase + row * ROW_STRIDE + ch;
    }
#pragma unroll
    for (int q = 0; q < 4; ++q) {
        int idx = tid + q * 256;
        int row = idx >> 3, ch = (idx & 7) * 16;
        gsrc[2 + q] = (const char*)Bm + ((size_t)(nBlock + row) * K) * 2 + ch;
        ldst[2 + q] = ldsBase + A_BUF_BYTES + row * ROW_STRIDE + ch;
    }

    v8f acc[4] = {};
    union Frag { v16bf v; struct { v8bf lo; v8bf hi; } p; };

    const int nk = K / KDEPTH;
    // prologue: stage K-block 0 into buffer 0
#pragma unroll
    for (int i = 0; i < 6; ++i) async_cp16(ldst[i], gsrc[i]);

    for (int ki = 0; ki < nk; ++ki) {
        const int cur = ki & 1;
        if (ki + 1 < nk) {
            const size_t kb = (size_t)(ki + 1) * SLAB_BYTES;  // bytes along K
            const uint32_t bo = (uint32_t)((cur ^ 1) * BUF_BYTES);
#pragma unroll
            for (int i = 0; i < 6; ++i) async_cp16(ldst[i] + bo, gsrc[i] + kb);
            asm volatile("s_wait_asynccnt 0x6" ::: "memory");
        } else {
            asm volatile("s_wait_asynccnt 0x0" ::: "memory");
        }
        __syncthreads();

        const unsigned char* sA = smem + cur * BUF_BYTES;
        const unsigned char* sB = sA + A_BUF_BYTES;
#pragma unroll
        for (int k2 = 0; k2 < 2; ++k2) {       // two 32-deep WMMA K-steps
            Frag a;
            const unsigned char* ar = sA + (wm * 16 + r16) * ROW_STRIDE + k2 * 64;
            a.p.lo = *(const v8bf*)(ar + g * 16);
            a.p.hi = *(const v8bf*)(ar + 32 + g * 16);
#pragma unroll
            for (int j = 0; j < 4; ++j) {
                const unsigned char* br = sB + (wn * 64 + j * 16 + r16) * ROW_STRIDE + k2 * 64;
                Frag b;
                b.p.lo = *(const v8bf*)(br + g * 16);
                b.p.hi = *(const v8bf*)(br + 32 + g * 16);
                acc[j] = __builtin_amdgcn_wmma_f32_16x16x32_bf16(
                    false, a.v, false, b.v, (short)0, acc[j], false, false);
            }
        }
        __syncthreads();
    }

#pragma unroll
    for (int j = 0; j < 4; ++j) {
        const int n    = nBlock + wn * 64 + j * 16 + r16;
        const int mOff = mBlock + wm * 16 + g * 8;
#pragma unroll
        for (int r = 0; r < 8; ++r) {
            size_t idx = (size_t)(mOff + r) * N + n;
            float v = acc[j][r];
            if (Res) v += Res[idx];
            C[idx] = v;
        }
    }
}

// ---------- causal depthwise conv1d + bias + SiLU ----------
__global__ void k_conv_silu(const float* __restrict__ xz, const float* __restrict__ cw,
                            const float* __restrict__ cb, float* __restrict__ u) {
    long i = (long)blockIdx.x * blockDim.x + threadIdx.x;  // over NTOK * D_INNER
    if (i >= (long)NTOK * D_INNER) return;
    int d   = (int)(i % D_INNER);
    int tok = (int)(i / D_INNER);
    int t   = tok % SEQLEN;
    float acc = cb[d];
#pragma unroll
    for (int k = 0; k < D_CONVK; ++k) {
        int tt = t - (D_CONVK - 1) + k;
        if (tt >= 0)
            acc += xz[(size_t)(tok - (D_CONVK - 1) + k) * (2 * D_INNER) + d] * cw[d * D_CONVK + k];
    }
    u[i] = siluf(acc);
}

// ---------- x_proj: proj[token, 0..95] = u_row . W_xproj_row ----------
__global__ __launch_bounds__(128)
void k_xproj(const float* __restrict__ u, const float* __restrict__ W,
             float* __restrict__ proj) {
    __shared__ float ur[D_INNER];
    const int tok = blockIdx.x;
    const float* up = u + (size_t)tok * D_INNER;
    for (int i = threadIdx.x; i < D_INNER; i += 128) ur[i] = up[i];
    __syncthreads();
    if (threadIdx.x < PROJ_W) {
        const float* wr = W + (size_t)threadIdx.x * D_INNER;
        float acc = 0.f;
#pragma unroll 8
        for (int k = 0; k < D_INNER; ++k) acc += ur[k] * wr[k];
        proj[(size_t)tok * PROJ_W + threadIdx.x] = acc;
    }
}

// ---------- dt_proj + softplus ----------
__global__ __launch_bounds__(256)
void k_dtproj(const float* __restrict__ proj, const float* __restrict__ Wdt,
              const float* __restrict__ bdt, float* __restrict__ dt) {
    __shared__ float p[DT_RANK];
    const int tok = blockIdx.x;
    if (threadIdx.x < DT_RANK) p[threadIdx.x] = proj[(size_t)tok * PROJ_W + threadIdx.x];
    __syncthreads();
    for (int d = threadIdx.x; d < D_INNER; d += 256) {
        const float* wr = Wdt + (size_t)d * DT_RANK;
        float acc = bdt[d];
#pragma unroll 8
        for (int r = 0; r < DT_RANK; ++r) acc += p[r] * wr[r];
        dt[(size_t)tok * D_INNER + d] = (acc > 20.f) ? acc : log1pf(__expf(acc));
    }
}

// ---------- selective scan + skip + gate, emit bf16 for out_proj ----------
__global__ __launch_bounds__(256)
void k_scan(const float* __restrict__ dtb, const float* __restrict__ u,
            const float* __restrict__ proj, const float* __restrict__ xz,
            const float* __restrict__ A_log, const float* __restrict__ Dv,
            bf16_t* __restrict__ gout) {
    const int gid = blockIdx.x * 256 + threadIdx.x;   // 0 .. B*D_INNER-1
    const int b = gid / D_INNER;
    const int d = gid % D_INNER;
    float a[D_STATE], s[D_STATE];
#pragma unroll
    for (int n = 0; n < D_STATE; ++n) {
        a[n] = -__expf(A_log[d * D_STATE + n]);
        s[n] = 0.f;
    }
    const float Dd = Dv[d];
    for (int t = 0; t < SEQLEN; ++t) {
        const size_t tok = (size_t)b * SEQLEN + t;
        const float dt = dtb[tok * D_INNER + d];
        const float uu = u[tok * D_INNER + d];
        const float* Br = proj + tok * PROJ_W + DT_RANK;
        const float* Cr = Br + D_STATE;
        const float du = dt * uu;
        float y = 0.f;
#pragma unroll
        for (int n = 0; n < D_STATE; ++n) {
            s[n] = __expf(dt * a[n]) * s[n] + du * Br[n];
            y += s[n] * Cr[n];
        }
        const float z = xz[tok * (2 * D_INNER) + D_INNER + d];
        gout[tok * D_INNER + d] = f2bf((y + uu * Dd) * siluf(z));
    }
}

extern "C" void kernel_launch(void* const* d_in, const int* in_sizes, int n_in,
                              void* d_out, int out_size, void* d_ws, size_t ws_size,
                              hipStream_t stream) {
    const float* x      = (const float*)d_in[0];
    const float* ln_g   = (const float*)d_in[1];
    const float* ln_b   = (const float*)d_in[2];
    const float* W_in   = (const float*)d_in[3];
    const float* conv_w = (const float*)d_in[4];
    const float* conv_b = (const float*)d_in[5];
    const float* W_xprj = (const float*)d_in[6];
    const float* W_dt   = (const float*)d_in[7];
    const float* b_dt   = (const float*)d_in[8];
    const float* A_log  = (const float*)d_in[9];
    const float* Dvec   = (const float*)d_in[10];
    const float* W_out  = (const float*)d_in[11];
    float* out = (float*)d_out;

    // workspace layout (256B aligned slabs)
    char* wsb = (char*)d_ws;
    size_t off = 0;
    auto alloc = [&](size_t bytes) -> void* {
        void* p = wsb + off;
        off += (bytes + 255) & ~(size_t)255;
        return p;
    };
    bf16_t* Wi_bf = (bf16_t*)alloc((size_t)(2 * D_INNER) * D_MODEL * 2); // 8.4 MB
    bf16_t* Wo_bf = (bf16_t*)alloc((size_t)D_MODEL * D_INNER * 2);       // 4.2 MB
    bf16_t* h_bf  = (bf16_t*)alloc((size_t)NTOK * D_MODEL * 2);          // 16.8 MB
    float*  xz    = (float*) alloc((size_t)NTOK * 2 * D_INNER * 4);      // 134 MB
    float*  u     = (float*) alloc((size_t)NTOK * D_INNER * 4);          // 67 MB
    float*  proj  = (float*) alloc((size_t)NTOK * PROJ_W * 4);           // 3.1 MB
    float*  dtb   = (float*) alloc((size_t)NTOK * D_INNER * 4);          // 67 MB
    bf16_t* g_bf  = (bf16_t*)alloc((size_t)NTOK * D_INNER * 2);          // 33.6 MB

    // 1) weights to bf16
    {
        int n1 = 2 * D_INNER * D_MODEL;
        k_cvt_bf16<<<(n1 + 255) / 256, 256, 0, stream>>>(W_in, Wi_bf, n1);
        int n2 = D_MODEL * D_INNER;
        k_cvt_bf16<<<(n2 + 255) / 256, 256, 0, stream>>>(W_out, Wo_bf, n2);
    }
    // 2) LayerNorm
    k_layernorm<<<NTOK, 256, 0, stream>>>(x, ln_g, ln_b, h_bf);
    // 3) in_proj GEMM: xz(8192x4096) = h(8192x1024) @ W_in(4096x1024)^T
    {
        dim3 grid((2 * D_INNER) / 128, NTOK / 64);
        k_gemm_bf16<<<grid, 256, 0, stream>>>(h_bf, Wi_bf, nullptr, xz,
                                              NTOK, 2 * D_INNER, D_MODEL);
    }
    // 4) causal depthwise conv + SiLU
    {
        long n = (long)NTOK * D_INNER;
        k_conv_silu<<<(unsigned)((n + 255) / 256), 256, 0, stream>>>(xz, conv_w, conv_b, u);
    }
    // 5) x_proj
    k_xproj<<<NTOK, 128, 0, stream>>>(u, W_xprj, proj);
    // 6) dt_proj + softplus
    k_dtproj<<<NTOK, 256, 0, stream>>>(proj, W_dt, b_dt, dtb);
    // 7) selective scan + skip + gate -> bf16
    k_scan<<<(BATCH * D_INNER) / 256, 256, 0, stream>>>(dtb, u, proj, xz, A_log, Dvec, g_bf);
    // 8) out_proj GEMM with fused residual: out = x + g(8192x2048) @ W_out(1024x2048)^T
    {
        dim3 grid(D_MODEL / 128, NTOK / 64);
        k_gemm_bf16<<<grid, 256, 0, stream>>>(g_bf, Wo_bf, x, out,
                                              NTOK, D_MODEL, D_INNER);
    }
    (void)in_sizes; (void)n_in; (void)out_size; (void)ws_size;
}
